// AGNNModel_35639638622732
// MI455X (gfx1250) — matrix operand order repaired and checked by reading, not compiled, not run on hardware
//
#include <hip/hip_runtime.h>
#include <hip/hip_bf16.h>
#include <math.h>

// ---------------------------------------------------------------------------
// AGNN (PyG AGNNConv x4 + Linear residual) for MI455X / gfx1250, wave32.
// Edge passes are L2-bandwidth bound (feature matrices fit in 192MB L2);
// the dense Linear uses v_wmma_f32_16x16x32_f16 with hi/lo f16 splitting
// for ~fp32 accuracy.
// ---------------------------------------------------------------------------

typedef __attribute__((ext_vector_type(16))) _Float16 v16h;
typedef __attribute__((ext_vector_type(8)))  float    v8f;

__device__ __forceinline__ v8f wmma_f16(v16h a, v16h b, v8f c) {
  return __builtin_amdgcn_wmma_f32_16x16x32_f16(
      /*neg_a=*/false, a, /*neg_b=*/false, b,
      /*c_mod=*/(short)0, c, /*reuse_a=*/false, /*reuse_b=*/false);
}

// ---------------------------------------------------------------------------
// res = x @ W^T + b   (N x 64) = (N x 64)(64 x 64)^T
// One wave per 16-row tile; K=64 -> 2 steps of K=32; 4 column tiles of 16.
// Split-f16: x = Ah + Al, W = Bh + Bl; accumulate AhBh + AhBl + AlBh in f32.
// ---------------------------------------------------------------------------
__global__ __launch_bounds__(256)
void linear_wmma_k(const float* __restrict__ x, const float* __restrict__ W,
                   const float* __restrict__ bias, float* __restrict__ res, int N) {
  const int lane = threadIdx.x & 31;
  const int wave = threadIdx.x >> 5;
  const long row0 = ((long)blockIdx.x * 8 + wave) * 16;
  if (row0 >= N) return;

  // ----- A fragments (16x32 f16 layout: lanes 0-15 kbase=0, 16-31 kbase=8;
  //       VGPR j: K=kbase+2j..+1, VGPR 4+j: K=kbase+16+2j..+1) -----
  const int m = lane & 15;
  long r = row0 + m;
  if (r >= N) r = (long)N - 1;               // clamp loads; stores are guarded
  const float* xr = x + r * 64;
  const int kb = (lane < 16) ? 0 : 8;

  v16h ah[2], al[2];
  #pragma unroll
  for (int ks = 0; ks < 2; ++ks) {
    #pragma unroll
    for (int j = 0; j < 4; ++j) {
      #pragma unroll
      for (int t = 0; t < 2; ++t) {
        float v0 = xr[ks * 32 + kb + 2 * j + t];
        float v1 = xr[ks * 32 + kb + 16 + 2 * j + t];
        _Float16 h0 = (_Float16)v0;
        _Float16 h1 = (_Float16)v1;
        ah[ks][2 * j + t]     = h0;
        al[ks][2 * j + t]     = (_Float16)(v0 - (float)h0);
        ah[ks][8 + 2 * j + t] = h1;
        al[ks][8 + 2 * j + t] = (_Float16)(v1 - (float)h1);
      }
    }
  }

  // ----- B fragments per (nt, ks): 32x16 f16 layout: lanes 0-15 kbase=0, 16-31 kbase=8;
  //       VGPR j: K=kbase+2j..+1, VGPR 4+j: K=kbase+16+2j..+1. B[k][n] = W[n][k]. -----
  const int n   = lane & 15;
  const int kb2 = (lane < 16) ? 0 : 8;

  #pragma unroll
  for (int nt = 0; nt < 4; ++nt) {
    v8f c = {0.f, 0.f, 0.f, 0.f, 0.f, 0.f, 0.f, 0.f};
    #pragma unroll
    for (int ks = 0; ks < 2; ++ks) {
      v16h bh, bl;
      const float* wr = W + (size_t)(nt * 16 + n) * 64 + ks * 32 + kb2;
      #pragma unroll
      for (int j = 0; j < 8; ++j) {
        #pragma unroll
        for (int t = 0; t < 2; ++t) {
          float wv = wr[2 * j + t + (j >= 4 ? 8 : 0)];
          _Float16 wh = (_Float16)wv;
          bh[2 * j + t] = wh;
          bl[2 * j + t] = (_Float16)(wv - (float)wh);
        }
      }
      c = wmma_f16(al[ks], bh, c);
      c = wmma_f16(ah[ks], bl, c);
      c = wmma_f16(ah[ks], bh, c);
    }
    // C/D layout: VGPR i -> M = i (+8 for lanes 16-31), N = lane&15
    const float bv = bias[nt * 16 + n];
    #pragma unroll
    for (int i = 0; i < 8; ++i) {
      long orow = row0 + i + ((lane >= 16) ? 8 : 0);
      if (orow < N) res[orow * 64 + nt * 16 + n] = c[i] + bv;
    }
  }
}

// ---------------------------------------------------------------------------
// Row L2-normalize: one wave32 per node, float2 per lane (256B coalesced row).
// ---------------------------------------------------------------------------
__global__ __launch_bounds__(256)
void normalize_k(const float* __restrict__ h, float* __restrict__ xn, int N) {
  const int lane = threadIdx.x & 31;
  const long node = (long)blockIdx.x * 8 + (threadIdx.x >> 5);
  if (node >= N) return;
  float2 v = ((const float2*)(h + node * 64))[lane];
  float ss = v.x * v.x + v.y * v.y;
  #pragma unroll
  for (int off = 16; off > 0; off >>= 1) ss += __shfl_xor(ss, off, 32);
  float inv = 1.0f / fmaxf(sqrtf(ss), 1e-12f);
  float2 o; o.x = v.x * inv; o.y = v.y * inv;
  ((float2*)(xn + node * 64))[lane] = o;
}

__global__ __launch_bounds__(256)
void init_amax_k(float* __restrict__ amax, int N) {
  long i = (long)blockIdx.x * blockDim.x + threadIdx.x;
  if (i < N) amax[i] = -3.402823466e38f;
}

// sign-aware float atomic max via integer atomics (init must be -FLT_MAX)
__device__ __forceinline__ void atomicMaxF(float* addr, float val) {
  if (val >= 0.0f)
    atomicMax((int*)addr, __float_as_int(val));
  else
    atomicMin((unsigned int*)addr, __float_as_uint(val));
}

// ---------------------------------------------------------------------------
// Pass 1: alpha[e] = beta * dot(xn[src], xn[dst]); segment max into amax[dst].
// One wave32 per edge; lanes hold float2 slices of both rows.
// ---------------------------------------------------------------------------
__global__ __launch_bounds__(256)
void edge_alpha_k(const float* __restrict__ xn, const int* __restrict__ src,
                  const int* __restrict__ dst, const float* __restrict__ betas,
                  int layer, float* __restrict__ alpha, float* __restrict__ amax,
                  long Etot, long E) {
  const int lane = threadIdx.x & 31;
  const long e = (long)blockIdx.x * 8 + (threadIdx.x >> 5);
  if (e >= Etot) return;
  int s, d;
  if (e < E) { s = src[e]; d = dst[e]; } else { s = d = (int)(e - E); }  // self-loop
  float2 a = ((const float2*)(xn + (long)s * 64))[lane];
  float2 b = ((const float2*)(xn + (long)d * 64))[lane];
  float p = a.x * b.x + a.y * b.y;
  #pragma unroll
  for (int off = 16; off > 0; off >>= 1) p += __shfl_xor(p, off, 32);
  if (lane == 0) {
    float al = betas[layer] * p;
    alpha[e] = al;
    atomicMaxF(&amax[d], al);
  }
}

// ---------------------------------------------------------------------------
// Pass 2: ex = exp(alpha - amax[dst]); segment sum into denom[dst]. 1 thread/edge.
// ---------------------------------------------------------------------------
__global__ __launch_bounds__(256)
void edge_exp_k(const int* __restrict__ dst, float* __restrict__ alpha,
                const float* __restrict__ amax, float* __restrict__ denom,
                long Etot, long E) {
  const long e = (long)blockIdx.x * blockDim.x + threadIdx.x;
  if (e >= Etot) return;
  const int d = (e < E) ? dst[e] : (int)(e - E);
  float ex = expf(alpha[e] - amax[d]);
  alpha[e] = ex;                                // overwrite in place
  atomicAdd(&denom[d], ex);
}

// ---------------------------------------------------------------------------
// Pass 3: out[dst] += (ex/denom[dst]) * h_in[src]. One wave32 per edge,
// float2 per lane -> 2 f32 atomic adds per lane into the 256B dst row.
// ---------------------------------------------------------------------------
__global__ __launch_bounds__(256)
void edge_scatter_k(const float* __restrict__ hin, const int* __restrict__ src,
                    const int* __restrict__ dst, const float* __restrict__ ex,
                    const float* __restrict__ denom, float* __restrict__ hout,
                    long Etot, long E) {
  const int lane = threadIdx.x & 31;
  const long e = (long)blockIdx.x * 8 + (threadIdx.x >> 5);
  if (e >= Etot) return;
  int s, d;
  if (e < E) { s = src[e]; d = dst[e]; } else { s = d = (int)(e - E); }
  const float w = ex[e] / denom[d];
  float2 v = ((const float2*)(hin + (long)s * 64))[lane];
  float* op = hout + (long)d * 64 + lane * 2;
  atomicAdd(op,     w * v.x);
  atomicAdd(op + 1, w * v.y);
}

__global__ __launch_bounds__(256)
void relu_k(float* __restrict__ h, long n) {
  long i = (long)blockIdx.x * blockDim.x + threadIdx.x;
  if (i < n) h[i] = fmaxf(h[i], 0.0f);
}

__global__ __launch_bounds__(256)
void addrelu_k(float* __restrict__ out, const float* __restrict__ res, long n) {
  long i = (long)blockIdx.x * blockDim.x + threadIdx.x;
  if (i < n) out[i] = fmaxf(out[i] + res[i], 0.0f);
}

// ---------------------------------------------------------------------------
extern "C" void kernel_launch(void* const* d_in, const int* in_sizes, int n_in,
                              void* d_out, int out_size, void* d_ws, size_t ws_size,
                              hipStream_t stream) {
  const float* x     = (const float*)d_in[0];   // [N,64]
  const int*   eidx  = (const int*)  d_in[1];   // [2,E]
  const float* W     = (const float*)d_in[2];   // [64,64]
  const float* bvec  = (const float*)d_in[3];   // [64]
  const float* betas = (const float*)d_in[4];   // [L]

  const int  N    = in_sizes[0] / 64;
  const long E    = in_sizes[1] / 2;
  const int  L    = in_sizes[4];
  const long Etot = E + N;                      // + self loops
  const int* src  = eidx;
  const int* dst  = eidx + E;

  const size_t feat = (size_t)N * 64;
  float* wsf   = (float*)d_ws;
  float* res   = wsf;                 // N*64
  float* xn    = res  + feat;         // N*64
  float* bufA  = xn   + feat;         // N*64
  float* bufB  = bufA + feat;         // N*64
  float* alpha = bufB + feat;         // E+N
  float* amax  = alpha + Etot;        // N
  float* denom = amax + N;            // N
  float* out   = (float*)d_out;

  // dense residual: res = x @ W^T + b  (WMMA)
  linear_wmma_k<<<(N + 127) / 128, 256, 0, stream>>>(x, W, bvec, res, N);

  auto conv = [&](const float* hin, float* hout, int layer) {
    normalize_k<<<(N + 7) / 8, 256, 0, stream>>>(hin, xn, N);
    init_amax_k<<<(N + 255) / 256, 256, 0, stream>>>(amax, N);
    hipMemsetAsync(denom, 0, (size_t)N * sizeof(float), stream);
    hipMemsetAsync(hout, 0, feat * sizeof(float), stream);
    edge_alpha_k<<<(int)((Etot + 7) / 8), 256, 0, stream>>>(
        xn, src, dst, betas, layer, alpha, amax, Etot, E);
    edge_exp_k<<<(int)((Etot + 255) / 256), 256, 0, stream>>>(
        dst, alpha, amax, denom, Etot, E);
    edge_scatter_k<<<(int)((Etot + 7) / 8), 256, 0, stream>>>(
        hin, src, dst, alpha, denom, hout, Etot, E);
  };

  // conv1 + relu
  conv(x, bufA, 0);
  relu_k<<<(int)((feat + 255) / 256), 256, 0, stream>>>(bufA, (long)feat);

  // middle convs (no activation), ping-pong buffers
  const float* cur = bufA;
  float* ping[2] = {bufB, bufA};
  int pi = 0;
  for (int li = 1; li < L - 1; ++li) {
    conv(cur, ping[pi], li);
    cur = ping[pi];
    pi ^= 1;
  }

  // conv_out -> d_out, then fused  out = relu(h + res)
  conv(cur, out, L - 1);
  addrelu_k<<<(int)((feat + 255) / 256), 256, 0, stream>>>(out, res, (long)feat);
}
